// GATLayer_67095979098970
// MI455X (gfx1250) — matrix operand order repaired and checked by reading, not compiled, not run on hardware
//
#include <hip/hip_runtime.h>

// GAT layer, fused flash-softmax over a dense 0/1 adjacency.
//   N=8192 nodes, IN=128, OUT=64, leaky slope 0.2
// Bottleneck: streaming A (256MB) once from HBM (~11us @ 23.3TB/s).
// Aggregation alpha@H and the softmax denominators both run on
// v_wmma_f32_16x16x32_f16 (5 WMMA per 16x32 tile: 4 N-tiles + ones-column).

#define NN      8192
#define IN_DIM  128
#define OUT_DIM 64
#define SLOPE   0.2f

typedef _Float16 v16h  __attribute__((ext_vector_type(16)));
typedef _Float16 half8 __attribute__((ext_vector_type(8)));
typedef float    v8f   __attribute__((ext_vector_type(8)));

// ---------------------------------------------------------------------------
// Kernel 1: one block (64 threads) per node row i.
//   H[i][t] = dot(X[i,:], W[:,t])         (fp32, kept in registers)
//   HT[t][i] = (f16) H[i][t]              (transposed f16 for WMMA B-operand)
//   e_l[i] = dot(H[i,:], a_l) ; e_r[i] = dot(H[i,:], a_r)
// ---------------------------------------------------------------------------
__global__ void gat_prep(const float* __restrict__ X, const float* __restrict__ W,
                         const float* __restrict__ al, const float* __restrict__ ar,
                         _Float16* __restrict__ HT, float* __restrict__ el,
                         float* __restrict__ er) {
  __shared__ float xs[IN_DIM];
  __shared__ float rl[OUT_DIM];
  __shared__ float rr[OUT_DIM];
  const int i = blockIdx.x;
  const int t = threadIdx.x;                 // 0..63 (output feature)
  xs[t]      = X[i * IN_DIM + t];
  xs[t + 64] = X[i * IN_DIM + t + 64];
  __syncthreads();
  float h = 0.f;
#pragma unroll 8
  for (int k = 0; k < IN_DIM; ++k) h += xs[k] * W[k * OUT_DIM + t];
  HT[(size_t)t * NN + i] = (_Float16)h;      // transposed f16 copy
  rl[t] = h * al[t];
  rr[t] = h * ar[t];
  __syncthreads();
  for (int s = 32; s > 0; s >>= 1) {
    if (t < s) { rl[t] += rl[t + s]; rr[t] += rr[t + s]; }
    __syncthreads();
  }
  if (t == 0) { el[i] = rl[0]; er[i] = rr[0]; }
}

// ---------------------------------------------------------------------------
// Kernel 2: one block (256 thr = 8 waves) per 16-row block of A.
// Wave w streams columns [w*1024, (w+1)*1024) in 16x32 tiles:
//   - lane builds its 16x16x32 f16 A-fragment of weights
//       w = A ? exp(leaky_relu(e_l[i]+e_r[j])) : 0
//     matching the ISA 16-bit A layout (lanes 0-15: K 0-7 & 16-23,
//     lanes 16-31: K 8-15 & 24-31)
//   - 4x wmma against HT (B layout matches A: lanes 0-15 K 0-7 & 16-23,
//     lanes 16-31 K 8-15 & 24-31, N = lane%16 -> two 16B reads per lane from HT)
//   - 1x wmma against an all-ones B fragment -> softmax denominators,
//     exactly consistent with the f16-rounded numerator weights
// Deterministic cross-wave combine through LDS, then divide by row sums.
// ---------------------------------------------------------------------------
__global__ void __launch_bounds__(256)
gat_attn(const int* __restrict__ A, const _Float16* __restrict__ HT,
         const float* __restrict__ el, const float* __restrict__ er,
         float* __restrict__ out) {
  __shared__ float ers[NN];                  // 32 KB: e_r staged once
  __shared__ float accs[8 * 16 * OUT_DIM];   // 32 KB: per-wave C tiles
  __shared__ float sums[8 * 16];             // per-wave row sums

  const int tid  = threadIdx.x;              // 0..255
  const int wave = tid >> 5;                 // 0..7 (K split)
  const int lane = tid & 31;
  const int half = lane >> 4;                // 0|1
  const int m    = lane & 15;                // row within 16-row block
  const int i0   = blockIdx.x * 16;

  // stage e_r -> LDS (coalesced b128)
  for (int k = tid; k < NN / 4; k += 256)
    ((float4*)ers)[k] = ((const float4*)er)[k];
  __syncthreads();

  const int   row   = i0 + m;
  const float eli   = el[row];
  const int*  arow  = A + (size_t)row * NN;
  const int   jbase = wave * 1024;

  union { v16h v; _Float16 h[16]; } onesu;
#pragma unroll
  for (int k = 0; k < 16; ++k) onesu.h[k] = (_Float16)1.0f;

  v8f c[4] = {};
  v8f csum = {};

  for (int tile = 0; tile < 32; ++tile) {
    const int j0 = jbase + tile * 32;
    const int k1 = j0 + half * 8;            // K 0-7 (lo half) / 8-15 (hi half)
    const int k2 = k1 + 16;                  // K 16-23 / 24-31
    // prefetch the A stream ahead (speculative, dropped if OOB)
    __builtin_prefetch((const void*)(arow + j0 + 512), 0, 1);

    union { int4 q[4]; int i[16]; } am;
    am.q[0] = *(const int4*)(arow + k1);
    am.q[1] = *(const int4*)(arow + k1 + 4);
    am.q[2] = *(const int4*)(arow + k2);
    am.q[3] = *(const int4*)(arow + k2 + 4);
    union { float4 q[4]; float f[16]; } ev;
    ev.q[0] = *(const float4*)(ers + k1);
    ev.q[1] = *(const float4*)(ers + k1 + 4);
    ev.q[2] = *(const float4*)(ers + k2);
    ev.q[3] = *(const float4*)(ers + k2 + 4);

    union { v16h v; _Float16 h[16]; } af;
#pragma unroll
    for (int k = 0; k < 16; ++k) {
      float x = eli + ev.f[k];
      float e = fmaxf(x, SLOPE * x);         // leaky_relu
      float w = am.i[k] ? __expf(e) : 0.f;   // mask non-edges
      af.h[k] = (_Float16)w;
    }

#pragma unroll
    for (int nt = 0; nt < 4; ++nt) {
      const _Float16* bp = HT + (size_t)(nt * 16 + m) * NN + j0 + half * 8;
      union { v16h v; half8 h[2]; } bf;
      bf.h[0] = *(const half8*)bp;
      bf.h[1] = *(const half8*)(bp + 16);
      c[nt] = __builtin_amdgcn_wmma_f32_16x16x32_f16(
          false, af.v, false, bf.v, (short)0, c[nt], false, false);
    }
    // denominator: row-sum of the same f16 weights via ones-column WMMA
    csum = __builtin_amdgcn_wmma_f32_16x16x32_f16(
        false, af.v, false, onesu.v, (short)0, csum, false, false);
  }

  // deterministic combine: each wave deposits its fragment + row sums
#pragma unroll
  for (int nt = 0; nt < 4; ++nt)
#pragma unroll
    for (int r = 0; r < 8; ++r) {
      int M = r + half * 8;                  // C/D layout: lanes>=16 hold M+8
      int Nc = nt * 16 + m;
      accs[wave * 1024 + M * OUT_DIM + Nc] = c[nt][r];
    }
  if (m == 0) {                              // lanes 0 & 16 cover rows 0-15
#pragma unroll
    for (int r = 0; r < 8; ++r)
      sums[wave * 16 + r + half * 8] = csum[r];
  }
  __syncthreads();

#pragma unroll
  for (int e = 0; e < 4; ++e) {
    int idx = e * 256 + tid;                 // 0..1023, coalesced stores
    int M = idx >> 6, Nc = idx & 63;
    float a = 0.f, s = 0.f;
#pragma unroll
    for (int w = 0; w < 8; ++w) {
      a += accs[w * 1024 + idx];
      s += sums[w * 16 + M];
    }
    out[(size_t)(i0 + M) * OUT_DIM + Nc] = a / s;
  }
}

// ---------------------------------------------------------------------------
extern "C" void kernel_launch(void* const* d_in, const int* in_sizes, int n_in,
                              void* d_out, int out_size, void* d_ws, size_t ws_size,
                              hipStream_t stream) {
  const float* X  = (const float*)d_in[0];
  const int*   A  = (const int*)  d_in[1];
  const float* W  = (const float*)d_in[2];
  const float* al = (const float*)d_in[3];
  const float* ar = (const float*)d_in[4];
  float* out = (float*)d_out;

  char* ws = (char*)d_ws;
  _Float16* HT = (_Float16*)ws;                                  // 1 MB
  float* el = (float*)(ws + (size_t)OUT_DIM * NN * 2);           // 32 KB
  float* er = (float*)(ws + (size_t)OUT_DIM * NN * 2 + NN * 4);  // 32 KB

  gat_prep<<<NN, 64, 0, stream>>>(X, W, al, ar, HT, el, er);
  gat_attn<<<NN / 16, 256, 0, stream>>>(A, HT, el, er, out);
}
